// MaskDecoderBlock_30150670418195
// MI455X (gfx1250) — matrix operand order repaired and checked by reading, not compiled, not run on hardware
//
#include <hip/hip_runtime.h>
#include <hip/hip_bf16.h>

typedef __attribute__((ext_vector_type(16))) _Float16 v16h;
typedef __attribute__((ext_vector_type(8)))  _Float16 h8v;
typedef __attribute__((ext_vector_type(4)))  _Float16 h4v;
typedef __attribute__((ext_vector_type(8)))  float    v8f;

#define WMMA_F16(a,b,c) __builtin_amdgcn_wmma_f32_16x16x32_f16(false,(a),false,(b),(short)0,(c),false,false)

// ---- CDNA5 async global->LDS (16B per lane), tracked by ASYNCcnt ----
__device__ __forceinline__ void async_g2l_b128(void* lds_dst, const void* gsrc) {
  unsigned l = (unsigned)(uintptr_t)lds_dst;            // LDS byte address = addr[31:0]
  unsigned long long g = (unsigned long long)(uintptr_t)gsrc;
  asm volatile("global_load_async_to_lds_b128 %0, %1, off"
               :: "v"(l), "v"(g) : "memory");
}
__device__ __forceinline__ void async_wait0() {
  asm volatile("s_wait_asynccnt 0x0" ::: "memory");
}

// ---- WMMA fragment loaders (contiguous k order, same for A and B) ----
// A 16x32 f16: lanes 0-15 -> M=lane, K 0..15; lanes 16-31 -> K 16..31
__device__ __forceinline__ v16h ld_fa(const _Float16* base, int ld) {
  const int lane = threadIdx.x & 31;
  const int r  = lane & 15;
  const int kh = (lane >> 4) << 4;            // 0 or 16
  const _Float16* p = base + r * ld + kh;
  h8v lo = *(const h8v*)(p);
  h8v hi = *(const h8v*)(p + 8);
  v16h a;
#pragma unroll
  for (int i = 0; i < 8; ++i) { a[i] = lo[i]; a[8 + i] = hi[i]; }
  return a;
}
// B 32x16 f16: lane n (0-15) holds column n, K 0..15; lanes 16-31 K 16..31. Column-contiguous source.
__device__ __forceinline__ v16h ld_fb(const _Float16* base, int ld) {
  const int lane = threadIdx.x & 31;
  const int n  = lane & 15;
  const int kh = (lane >> 4) << 4;            // 0 or 16
  const _Float16* p = base + n * ld + kh;
  h8v lo = *(const h8v*)(p);
  h8v hi = *(const h8v*)(p + 8);
  v16h b;
#pragma unroll
  for (int i = 0; i < 8; ++i) { b[i] = lo[i]; b[8 + i] = hi[i]; }
  return b;
}
// B operand loaded straight from f32 weights in global (L2-hot), converting to f16.
__device__ __forceinline__ v16h ld_fb_f32(const float* base, int ld) {
  const int lane = threadIdx.x & 31;
  const int n  = lane & 15;
  const int kh = (lane >> 4) << 4;
  const float* p = base + n * ld + kh;
  v16h b;
#pragma unroll
  for (int i = 0; i < 16; ++i) b[i] = (_Float16)p[i];
  return b;
}

// ===================== Kernel 1: f32 GEMM -> f16 (K/V fused proj, Q proj) ==================
// out[M x N] = scale*A[M x 512] @ [Wa;Wb]^T, f16 output. BLOCK_M=128, full A K-panel in LDS.
// (f32 -> f16 conversion prevents async staging here; prefetch next W tile instead.)
__global__ void __launch_bounds__(256) proj_gemm_k(
    const float* __restrict__ A, const float* __restrict__ Wa,
    const float* __restrict__ Wb, _Float16* __restrict__ out,
    int M, int N, int N1, float scale)
{
  extern __shared__ char smem[];
  _Float16* As = (_Float16*)smem;                 // 128 x 520
  _Float16* Ws = As + 128 * 520;                  // 64 x 520
  const int tid  = threadIdx.x;
  const int wave = tid >> 5;
  const int lane = tid & 31;
  const int mbase = blockIdx.x * 128;

  // stage A rows once (f32 -> f16, scaled): feats read exactly once from HBM
  for (int i = tid; i < 128 * 128; i += 256) {
    int r = i >> 7, c4 = i & 127;
    int gr = mbase + r;
    float4 v = {0.f, 0.f, 0.f, 0.f};
    if (gr < M) v = ((const float4*)A)[(size_t)gr * 128 + c4];
    _Float16* d = As + r * 520 + c4 * 4;
    d[0] = (_Float16)(v.x * scale); d[1] = (_Float16)(v.y * scale);
    d[2] = (_Float16)(v.z * scale); d[3] = (_Float16)(v.w * scale);
  }

  const int nch = N >> 6;
  for (int nc = 0; nc < nch; ++nc) {
    __syncthreads();
    for (int i = tid; i < 64 * 128; i += 256) {       // stage 64 weight rows (f32->f16)
      int r = i >> 7, c4 = i & 127;
      int j = nc * 64 + r;
      const float* src = (j < N1) ? (Wa + (size_t)j * 512) : (Wb + (size_t)(j - N1) * 512);
      float4 v = ((const float4*)src)[c4];
      _Float16* d = Ws + r * 520 + c4 * 4;
      d[0]=(_Float16)v.x; d[1]=(_Float16)v.y; d[2]=(_Float16)v.z; d[3]=(_Float16)v.w;
    }
    __syncthreads();

    // prefetch next weight tile into cache while we compute (global_prefetch_b8)
    if (nc + 1 < nch) {
      int j = (nc + 1) * 64 + (tid & 63);
      const float* src = (j < N1) ? (Wa + (size_t)j * 512) : (Wb + (size_t)(j - N1) * 512);
      __builtin_prefetch(src + (tid >> 6) * 128, 0, 0);
    }

    v8f acc[4] = {};
    const _Float16* Ab = As + wave * 16 * 520;        // wave owns 16 M-rows
    for (int kk = 0; kk < 16; ++kk) {
      v16h af = ld_fa(Ab + kk * 32, 520);
#pragma unroll
      for (int t = 0; t < 4; ++t) {
        v16h bf = ld_fb(Ws + t * 16 * 520 + kk * 32, 520);
        acc[t] = WMMA_F16(af, bf, acc[t]);
      }
    }
    const int n = lane & 15, mo = (lane >> 4) * 8;
#pragma unroll
    for (int t = 0; t < 4; ++t) {
      int gn = nc * 64 + t * 16 + n;
#pragma unroll
      for (int v = 0; v < 8; ++v) {
        int gm = mbase + wave * 16 + mo + v;
        if (gm < M) out[(size_t)gm * N + gn] = (_Float16)acc[t][v];
      }
    }
  }
}

// ===================== Kernel 2: mask = Qscaled @ K^T (full 512-dim dot == sum over heads) ===
__global__ void __launch_bounds__(256) mask_gemm_k(
    const _Float16* __restrict__ qh, const _Float16* __restrict__ kv,
    float* __restrict__ mask)
{
  extern __shared__ char smem[];
  _Float16* Aq = (_Float16*)smem;         // 128 x 72
  _Float16* Bk = Aq + 128 * 72;           // 128 x 72
  const int tid = threadIdx.x;
  const int bb = blockIdx.x >> 7;         // batch
  const int nt = blockIdx.x & 127;        // 128-key tile
  const int wave = tid >> 5;
  const int wm = wave >> 1, wn = wave & 1;
  const int lane = tid & 31;
  v8f acc[2][4] = {};

  // zero the padded A rows (100..127) once; async staging below never writes them
  for (int i = tid; i < 28 * 16; i += 256) {
    int r = 100 + (i >> 4), c4 = i & 15;
    h4v z = {};
    *(h4v*)(Aq + r * 72 + c4 * 4) = z;
  }

  for (int kc = 0; kc < 8; ++kc) {        // K = 512 in chunks of 64
    __syncthreads();
    // async stage A: 100 rows x 64 cols (8 x b128 units per row)
    for (int i = tid; i < 128 * 8; i += 256) {
      int r = i >> 3, u = i & 7;
      if (r < 100)
        async_g2l_b128(Aq + r * 72 + u * 8,
                       qh + (size_t)(bb * 100 + r) * 512 + kc * 64 + u * 8);
    }
    // async stage B: 128 key rows x 64 cols
    for (int i = tid; i < 128 * 8; i += 256) {
      int r = i >> 3, u = i & 7;
      size_t row = (size_t)bb * 16384 + nt * 128 + r;
      async_g2l_b128(Bk + r * 72 + u * 8,
                     kv + row * 1024 + kc * 64 + u * 8);
    }
    async_wait0();
    __syncthreads();
#pragma unroll
    for (int kk = 0; kk < 2; ++kk) {
      v16h af[2], bf[4];
#pragma unroll
      for (int im = 0; im < 2; ++im) af[im] = ld_fa(Aq + (wm * 32 + im * 16) * 72 + kk * 32, 72);
#pragma unroll
      for (int t = 0; t < 4; ++t) bf[t] = ld_fb(Bk + (wn * 64 + t * 16) * 72 + kk * 32, 72);
#pragma unroll
      for (int im = 0; im < 2; ++im)
#pragma unroll
        for (int t = 0; t < 4; ++t)
          acc[im][t] = WMMA_F16(af[im], bf[t], acc[im][t]);
    }
  }
  const int n = lane & 15, mo = (lane >> 4) * 8;
#pragma unroll
  for (int im = 0; im < 2; ++im)
#pragma unroll
    for (int t = 0; t < 4; ++t)
#pragma unroll
      for (int v = 0; v < 8; ++v) {
        int l = wm * 32 + im * 16 + mo + v;
        if (l < 100) {
          int col = nt * 128 + wn * 64 + t * 16 + n;
          mask[((size_t)bb * 100 + l) * 16384 + col] = acc[im][t][v];
        }
      }
}

// ===================== Kernel 3: flash attention partials per (split,b,h) ===================
__global__ void __launch_bounds__(256) flash_part_k(
    const _Float16* __restrict__ qh, const _Float16* __restrict__ kv,
    float* __restrict__ Opart, float* __restrict__ MSpart)
{
  extern __shared__ char smem[];
  _Float16* qs  = (_Float16*)smem;          // 128 x 72 f16
  _Float16* Pls = qs + 128 * 72;            // 128 x 72 f16
  _Float16* vT  = Pls + 128 * 72;           // 64 x 72 f16 (V transposed: [d][m])
  float* Sls  = (float*)(vT + 64 * 72);     // 128 x 72 f32
  float* Mrow = Sls + 128 * 72;             // running max
  float* Srow = Mrow + 128;                 // running sum
  float* Arow = Srow + 128;                 // rescale alpha

  const int tid = threadIdx.x;
  const int h  = blockIdx.x & 7;
  const int bb = (blockIdx.x >> 3) & 15;
  const int split = blockIdx.x >> 7;
  const int wave = tid >> 5;
  const int lane = tid & 31;
  const int n = lane & 15, mo = (lane >> 4) * 8;

  // zero pad rows, then async-stage the 100 valid q rows (head slice)
  for (int i = tid; i < 28 * 16; i += 256) {
    int r = 100 + (i >> 4), c4 = i & 15;
    h4v z = {};
    *(h4v*)(qs + r * 72 + c4 * 4) = z;
  }
  for (int i = tid; i < 128 * 8; i += 256) {
    int r = i >> 3, u = i & 7;
    if (r < 100)
      async_g2l_b128(qs + r * 72 + u * 8,
                     qh + (size_t)(bb * 100 + r) * 512 + h * 64 + u * 8);
  }
  if (tid < 128) { Mrow[tid] = -1e30f; Srow[tid] = 0.f; }
  v8f accO[4] = {};
  async_wait0();
  __syncthreads();

  const size_t krow0 = ((size_t)bb * 16384 + (size_t)split * 4096) * 1024;
  for (int mc = 0; mc < 64; ++mc) {             // 64 keys per step, 4096 per split
    __syncthreads();
    for (int i = tid; i < 64 * 64; i += 256) {  // stage V chunk transposed (scatter: VALU path)
      int m = i >> 6, d = i & 63;
      vT[d * 72 + m] = kv[krow0 + (size_t)(mc * 64 + m) * 1024 + 512 + h * 64 + d];
    }
    // S = q . k^T  (B-operand straight from global kv, column = key row, d-contiguous)
    v8f accS[4] = {};
#pragma unroll
    for (int kk = 0; kk < 2; ++kk) {
      v16h af = ld_fa(qs + wave * 16 * 72 + kk * 32, 72);
#pragma unroll
      for (int t = 0; t < 4; ++t) {
        const _Float16* kp = kv + krow0 + (size_t)(mc * 64 + t * 16) * 1024 + h * 64 + kk * 32;
        v16h bf = ld_fb(kp, 1024);
        accS[t] = WMMA_F16(af, bf, accS[t]);
      }
    }
#pragma unroll
    for (int t = 0; t < 4; ++t)
#pragma unroll
      for (int v = 0; v < 8; ++v)
        Sls[(wave * 16 + mo + v) * 72 + t * 16 + n] = accS[t][v];
    __syncthreads();

    if (tid < 128) {                            // online softmax per row
      int r = tid;
      float mx = -1e30f;
      for (int j = 0; j < 64; ++j) mx = fmaxf(mx, Sls[r * 72 + j]);
      float Mn = fmaxf(Mrow[r], mx);
      float al = __expf(Mrow[r] - Mn);
      float s = 0.f;
      for (int j = 0; j < 64; ++j) {
        float p = __expf(Sls[r * 72 + j] - Mn);
        Pls[r * 72 + j] = (_Float16)p;
        s += p;
      }
      Srow[r] = Srow[r] * al + s;
      Mrow[r] = Mn;
      Arow[r] = al;
    }
    __syncthreads();

    float alv[8];
#pragma unroll
    for (int v = 0; v < 8; ++v) alv[v] = Arow[wave * 16 + mo + v];
#pragma unroll
    for (int t = 0; t < 4; ++t)
#pragma unroll
      for (int v = 0; v < 8; ++v) accO[t][v] *= alv[v];
#pragma unroll
    for (int kk = 0; kk < 2; ++kk) {            // O += P @ V
      v16h af = ld_fa(Pls + wave * 16 * 72 + kk * 32, 72);
#pragma unroll
      for (int t = 0; t < 4; ++t) {
        v16h bf = ld_fb(vT + t * 16 * 72 + kk * 32, 72);
        accO[t] = WMMA_F16(af, bf, accO[t]);
      }
    }
  }
  const size_t obase = (size_t)blockIdx.x * 128 * 64;
#pragma unroll
  for (int t = 0; t < 4; ++t)
#pragma unroll
    for (int v = 0; v < 8; ++v)
      Opart[obase + (size_t)(wave * 16 + mo + v) * 64 + t * 16 + n] = accO[t][v];
  if (tid < 128) {
    size_t msb = ((size_t)blockIdx.x * 128 + tid) * 2;
    MSpart[msb] = Mrow[tid]; MSpart[msb + 1] = Srow[tid];
  }
}

// ===================== Kernel 4: combine flash splits ======================================
__global__ void __launch_bounds__(256) flash_reduce_k(
    const float* __restrict__ Opart, const float* __restrict__ MSpart,
    _Float16* __restrict__ attnout)
{
  const int h = blockIdx.x & 7;
  const int bb = blockIdx.x >> 3;
  for (int i = threadIdx.x; i < 100 * 64; i += 256) {
    int r = i >> 6, d = i & 63;
    float Ms[4], Ss[4], Mx = -1e30f;
#pragma unroll
    for (int s = 0; s < 4; ++s) {
      size_t base = ((size_t)(s * 128 + bb * 8 + h) * 128 + r) * 2;
      Ms[s] = MSpart[base]; Ss[s] = MSpart[base + 1];
      Mx = fmaxf(Mx, Ms[s]);
    }
    float T = 0.f, o = 0.f;
#pragma unroll
    for (int s = 0; s < 4; ++s) {
      float w = __expf(Ms[s] - Mx);
      T += w * Ss[s];
      o += w * Opart[((size_t)(s * 128 + bb * 8 + h) * 128 + r) * 64 + d];
    }
    attnout[((size_t)(bb * 100 + r)) * 512 + h * 64 + d] = (_Float16)(o / T);
  }
}

// ===================== Kernel 5: 64-row GEMM + (residual+bias+LN) or (bias+ReLU) epilogue ===
__global__ void __launch_bounds__(256) tail_gemm_k(
    const _Float16* __restrict__ Ain, const float* __restrict__ W,
    const float* __restrict__ bias, const float* __restrict__ resid,
    const float* __restrict__ lnw, const float* __restrict__ lnb,
    _Float16* __restrict__ out16, float* __restrict__ out32, int do_relu)
{
  extern __shared__ char smem[];
  _Float16* As = (_Float16*)smem;           // 64 x 520 f16
  float* Xb = (float*)(As + 64 * 520);      // 64 x 520 f32
  const int tid = threadIdx.x;
  const int wave = tid >> 5;
  const int lane = tid & 31;
  const int mb = blockIdx.x * 64;

  // async stage the 64x512 f16 A panel (64 b128 units per row)
  for (int i = tid; i < 64 * 64; i += 256) {
    int r = i >> 6, u = i & 63;
    async_g2l_b128(As + r * 520 + u * 8,
                   Ain + (size_t)(mb + r) * 512 + u * 8);
  }
  async_wait0();
  __syncthreads();

  v8f acc[4][4] = {};                       // 64 rows x 64 cols per wave (wave owns N-chunk)
  for (int kk = 0; kk < 16; ++kk) {
    v16h af[4];
#pragma unroll
    for (int mt = 0; mt < 4; ++mt) af[mt] = ld_fa(As + mt * 16 * 520 + kk * 32, 520);
#pragma unroll
    for (int t = 0; t < 4; ++t) {
      v16h bf = ld_fb_f32(W + (size_t)(wave * 64 + t * 16) * 512 + kk * 32, 512);
#pragma unroll
      for (int mt = 0; mt < 4; ++mt)
        acc[mt][t] = WMMA_F16(af[mt], bf, acc[mt][t]);
    }
  }
  const int n = lane & 15, mo = (lane >> 4) * 8;
#pragma unroll
  for (int mt = 0; mt < 4; ++mt)
#pragma unroll
    for (int t = 0; t < 4; ++t)
#pragma unroll
      for (int v = 0; v < 8; ++v)
        Xb[(mt * 16 + mo + v) * 520 + wave * 64 + t * 16 + n] = acc[mt][t][v];
  __syncthreads();

  if (do_relu) {
    for (int i = tid; i < 64 * 512; i += 256) {
      int r = i >> 9, c = i & 511;
      float v = fmaxf(Xb[r * 520 + c] + bias[c], 0.f);
      out16[(size_t)(mb + r) * 512 + c] = (_Float16)v;
    }
  } else {
    if (tid < 64) {
      int r = tid;
      float sum = 0.f, ssq = 0.f;
      for (int c = 0; c < 512; ++c) {
        float v = Xb[r * 520 + c] + bias[c] + resid[(size_t)(mb + r) * 512 + c];
        Xb[r * 520 + c] = v;
        sum += v; ssq += v * v;
      }
      float mu = sum * (1.f / 512.f);
      float var = ssq * (1.f / 512.f) - mu * mu;
      float rs = rsqrtf(var + 1e-5f);
      for (int c = 0; c < 512; ++c) {
        float y = (Xb[r * 520 + c] - mu) * rs * lnw[c] + lnb[c];
        if (out32) out32[(size_t)(mb + r) * 512 + c] = y;
        if (out16) out16[(size_t)(mb + r) * 512 + c] = (_Float16)y;
      }
    }
  }
}

extern "C" void kernel_launch(void* const* d_in, const int* in_sizes, int n_in,
                              void* d_out, int out_size, void* d_ws, size_t ws_size,
                              hipStream_t stream) {
  (void)in_sizes; (void)n_in; (void)out_size; (void)ws_size;
  const float* query = (const float*)d_in[0];
  const float* feats = (const float*)d_in[1];
  const float* Wq = (const float*)d_in[2];
  const float* Wk = (const float*)d_in[3];
  const float* Wv = (const float*)d_in[4];
  const float* Wp = (const float*)d_in[5];
  const float* bp = (const float*)d_in[6];
  const float* W1 = (const float*)d_in[7];
  const float* b1 = (const float*)d_in[8];
  const float* W2 = (const float*)d_in[9];
  const float* b2 = (const float*)d_in[10];
  const float* ln0w = (const float*)d_in[11];
  const float* ln0b = (const float*)d_in[12];
  const float* ln2w = (const float*)d_in[13];
  const float* ln2b = (const float*)d_in[14];

  char* ws = (char*)d_ws;
  size_t off = 0;
  _Float16* kv      = (_Float16*)(ws + off); off += (size_t)262144 * 1024 * 2; // K|V f16
  _Float16* qh      = (_Float16*)(ws + off); off += (size_t)1600 * 512 * 2;    // scaled Q f16
  _Float16* attnout = (_Float16*)(ws + off); off += (size_t)1600 * 512 * 2;
  _Float16* x16     = (_Float16*)(ws + off); off += (size_t)1600 * 512 * 2;
  _Float16* hid16   = (_Float16*)(ws + off); off += (size_t)1600 * 512 * 2;
  float*    x32     = (float*)(ws + off);    off += (size_t)1600 * 512 * 4;
  float*    Opart   = (float*)(ws + off);    off += (size_t)512 * 128 * 64 * 4;
  float*    MSp     = (float*)(ws + off);    off += (size_t)512 * 128 * 2 * 4;

  float* y = (float*)d_out;
  float* mask = y + (size_t)16 * 100 * 512;

  const int SM_PROJ  = (128 * 520 + 64 * 520) * 2;                     // 199,680 B
  const int SM_MASK  = (128 * 72 + 128 * 72) * 2;                      //  36,864 B
  const int SM_FLASH = (128 * 72 + 128 * 72 + 64 * 72) * 2
                     + (128 * 72 + 3 * 128) * 4;                       //  84,480 B
  const int SM_TAIL  = 64 * 520 * 2 + 64 * 520 * 4;                    // 199,680 B

  // 1) K|V projection (feats read once; 275 GFLOP of wmma)
  proj_gemm_k<<<2048, 256, SM_PROJ, stream>>>(feats, Wk, Wv, kv, 262144, 1024, 512, 1.0f);
  // 2) Q projection with 1/sqrt(64) scale
  proj_gemm_k<<<13, 256, SM_PROJ, stream>>>(query, Wq, Wq, qh, 1600, 512, 512, 0.125f);
  // 3) mask = sum_h raw scores == full-dim Qscaled @ K^T
  mask_gemm_k<<<2048, 256, SM_MASK, stream>>>(qh, kv, mask);
  // 4) flash attention partials (4 key-splits x 16 batches x 8 heads)
  flash_part_k<<<512, 256, SM_FLASH, stream>>>(qh, kv, Opart, MSp);
  // 5) combine splits
  flash_reduce_k<<<128, 256, 0, stream>>>(Opart, MSp, attnout);
  // 6) x = LN(query + attnout @ Wp^T + bp)
  tail_gemm_k<<<25, 256, SM_TAIL, stream>>>(attnout, Wp, bp, query, ln0w, ln0b, x16, x32, 0);
  // 7) h = relu(x @ W1^T + b1)
  tail_gemm_k<<<25, 256, SM_TAIL, stream>>>(x16, W1, b1, nullptr, nullptr, nullptr, hid16, nullptr, 1);
  // 8) y = LN(x + h @ W2^T + b2)
  tail_gemm_k<<<25, 256, SM_TAIL, stream>>>(hid16, W2, b2, x32, ln2w, ln2b, nullptr, y, 0);
}
